// EulerRotaryAttention_40991167873346
// MI455X (gfx1250) — compile-verified, matrix operand order checked by reading, not run on hardware
//
#include <hip/hip_runtime.h>
#include <hip/hip_bf16.h>
#include <math.h>

// ---------------------------------------------------------------------------
// Problem constants (from setup_inputs: b=2, n=2048, d=1024, H=16)
// ---------------------------------------------------------------------------
#define B_    2
#define N_    2048
#define D_    1024
#define H_    16
#define DH_   64
#define M_TOT (B_*N_)   // 4096 tokens total
#define N3_   (3*D_)    // 3072 qkv columns

typedef __bf16 bf16_t;
typedef bf16_t v16bf __attribute__((ext_vector_type(16)));
typedef bf16_t v8bf  __attribute__((ext_vector_type(8)));
typedef float  v8f   __attribute__((ext_vector_type(8)));

__device__ inline v8f wmma_bf16(v16bf a, v16bf b, v8f c) {
  // D = A(16x32 bf16) * B(32x16 bf16) + C(16x16 f32)
  return __builtin_amdgcn_wmma_f32_16x16x32_bf16(
      /*neg_a=*/false, a, /*neg_b=*/false, b,
      /*c_mod=*/(short)0, c, /*reuse_a=*/false, /*reuse_b=*/false);
}

__device__ inline v8f zero8() {
  v8f z = {0.f, 0.f, 0.f, 0.f, 0.f, 0.f, 0.f, 0.f};
  return z;
}

// Assemble a 16x32 bf16 A-fragment (or 32x16 B-fragment; same lane mapping)
// from two contiguous 8-element (16B) runs. Per ISA 7.12.2: for 16-bit A,
// lane L holds row m=L%16; element e maps to K = (e<8 ? 8*(L/16)+e
//                                                     : 16+8*(L/16)+(e-8)).
__device__ inline v16bf make_frag(const bf16_t* p0, const bf16_t* p1) {
  v8bf lo = *(const v8bf*)p0;
  v8bf hi = *(const v8bf*)p1;
  v16bf f;
#pragma unroll
  for (int i = 0; i < 8; ++i) { f[i] = lo[i]; f[i + 8] = hi[i]; }
  return f;
}

// ---------------------------------------------------------------------------
// Prep kernels
// ---------------------------------------------------------------------------
__global__ void f32_to_bf16_kernel(const float* __restrict__ src,
                                   bf16_t* __restrict__ dst, int n) {
  int i = blockIdx.x * blockDim.x + threadIdx.x;
  if (i < n) dst[i] = (bf16_t)src[i];
}

// src (rows x cols) row-major fp32 -> dst (cols x rows) row-major bf16
__global__ void transpose_to_bf16_kernel(const float* __restrict__ src,
                                         bf16_t* __restrict__ dst,
                                         int rows, int cols) {
  int i = blockIdx.x * blockDim.x + threadIdx.x;
  if (i < rows * cols) {
    int r = i / cols, c = i - r * cols;
    dst[(size_t)c * rows + r] = (bf16_t)src[i];
  }
}

// cos/sin tables: [h][pos][j], j = 0..31 pair index within head dim.
// freqs[h] = sigmoid(bit_logits[h]) . (1, pi, e, pi*e, pi/e)
// theta    = pos * freqs[h] * 2^(-(2j)/64)
__global__ void euler_cossin_kernel(const float* __restrict__ bit_logits,
                                    float* __restrict__ cosT,
                                    float* __restrict__ sinT) {
  int idx = blockIdx.x * blockDim.x + threadIdx.x;
  if (idx >= H_ * N_ * 32) return;
  int j   = idx & 31;
  int pos = (idx >> 5) & (N_ - 1);
  int h   = idx >> 16;  // 32*2048 = 65536 entries per head
  const float PI = 3.14159265358979323846f;
  const float E  = 2.71828182845904523536f;
  float f = 0.f;
  f += 1.0f     / (1.f + __expf(-bit_logits[h * 5 + 0]));
  f += PI       / (1.f + __expf(-bit_logits[h * 5 + 1]));
  f += E        / (1.f + __expf(-bit_logits[h * 5 + 2]));
  f += (PI * E) / (1.f + __expf(-bit_logits[h * 5 + 3]));
  f += (PI / E) / (1.f + __expf(-bit_logits[h * 5 + 4]));
  float invf  = exp2f(-((2.0f * (float)j) / 64.0f));
  float theta = (float)pos * f * invf;
  float s, c;
  __sincosf(theta, &s, &c);
  cosT[idx] = c;
  sinT[idx] = s;
}

// ---------------------------------------------------------------------------
// Tiled bf16 WMMA GEMM:  C[M][Nn] = A[M][K] * Bt[Nn][K]^T
// Block tile 128x128, BK=32, 8 waves (2 M x 4 N), each wave 64x32 (4x2 tiles).
// EPI==0: QKV epilogue (fused RoPE via shfl_xor partner; scatter q/k/v).
// EPI==1: plain fp32 store to outF.
// ---------------------------------------------------------------------------
template <int EPI>
__global__ __launch_bounds__(256) void gemm_wmma_kernel(
    const bf16_t* __restrict__ A, const bf16_t* __restrict__ Bt,
    int M, int Nn, int K,
    float* __restrict__ outF,
    bf16_t* __restrict__ qb, bf16_t* __restrict__ kb, bf16_t* __restrict__ vT,
    const float* __restrict__ cosT, const float* __restrict__ sinT) {
  (void)M;
  __shared__ __align__(16) bf16_t As[128 * 32];
  __shared__ __align__(16) bf16_t Bs[128 * 32];

  const int tid  = threadIdx.x;
  const int lane = tid & 31, wave = tid >> 5;
  const int lm   = lane & 15;
  const int half = lane >> 4;
  const int h8   = half * 8;
  const int waveM = wave >> 2;   // 0..1  (64 rows each)
  const int waveN = wave & 3;    // 0..3  (32 cols each)
  const int mb = blockIdx.y * 128, nb = blockIdx.x * 128;

  v8f acc[4][2];
#pragma unroll
  for (int i = 0; i < 4; ++i)
#pragma unroll
    for (int j = 0; j < 2; ++j) acc[i][j] = zero8();

  for (int k0 = 0; k0 < K; k0 += 32) {
    // Stage 128x32 bf16 panels of A and Bt into LDS (16B per thread per panel
    // per pass; 2 passes each).
#pragma unroll
    for (int i = 0; i < 2; ++i) {
      int seg = tid + i * 256;
      int row = seg >> 2, cs = (seg & 3) * 8;
      *(v8bf*)&As[row * 32 + cs] =
          *(const v8bf*)&A[(size_t)(mb + row) * K + k0 + cs];
      *(v8bf*)&Bs[row * 32 + cs] =
          *(const v8bf*)&Bt[(size_t)(nb + row) * K + k0 + cs];
    }
    __syncthreads();

    v16bf af[4], bfr[2];
#pragma unroll
    for (int mi = 0; mi < 4; ++mi) {
      int r = waveM * 64 + mi * 16 + lm;
      af[mi] = make_frag(&As[r * 32 + h8], &As[r * 32 + 16 + h8]);
    }
#pragma unroll
    for (int ni = 0; ni < 2; ++ni) {
      int r = waveN * 32 + ni * 16 + lm;
      bfr[ni] = make_frag(&Bs[r * 32 + h8], &Bs[r * 32 + 16 + h8]);
    }
#pragma unroll
    for (int mi = 0; mi < 4; ++mi)
#pragma unroll
      for (int ni = 0; ni < 2; ++ni)
        acc[mi][ni] = wmma_bf16(af[mi], bfr[ni], acc[mi][ni]);
    __syncthreads();
  }

  // Epilogue. C layout: lane holds col n = lm, rows m = r + 8*half (r=0..7).
#pragma unroll
  for (int mi = 0; mi < 4; ++mi) {
#pragma unroll
    for (int ni = 0; ni < 2; ++ni) {
#pragma unroll
      for (int r = 0; r < 8; ++r) {
        int grow = mb + waveM * 64 + mi * 16 + r + 8 * half;  // token index
        int gcol = nb + waveN * 32 + ni * 16 + lm;            // output column
        float val = acc[mi][ni][r];
        if (EPI == 1) {
          outF[(size_t)grow * Nn + gcol] = val;
        } else {
          // RoPE partner value lives in lane^1 (adjacent column).
          float partner = __shfl_xor(val, 1);
          int which = gcol >> 10;       // 0=q 1=k 2=v
          int cd  = gcol & 1023;
          int hh  = cd >> 6;            // head
          int col = cd & 63;            // within-head dim
          int j   = col >> 1;           // rotation pair index
          int bidx = grow >> 11;        // batch
          int pos  = grow & 2047;       // sequence position
          int bh   = bidx * H_ + hh;
          if (which < 2) {
            float c = cosT[((size_t)hh * N_ + pos) * 32 + j];
            float s = sinT[((size_t)hh * N_ + pos) * 32 + j];
            float sgn = (gcol & 1) ? 1.0f : -1.0f;  // rx1=-x2*s, rx2=+x1*s
            float rot = val * c + sgn * partner * s;
            bf16_t* dst = (which == 0 ? qb : kb);
            dst[((size_t)bh * N_ + pos) * DH_ + col] = (bf16_t)rot;
          } else {
            // v stored transposed [b,h,dh,n] so PV B-fragments are contiguous
            vT[((size_t)bh * DH_ + col) * N_ + pos] = (bf16_t)val;
          }
        }
      }
    }
  }
}

// ---------------------------------------------------------------------------
// Flash attention with causal mask. One wave = 16 query rows; block = 4 waves
// = 64 query rows; grid = (n/64, b*H). Online softmax in fp32; S=QK^T and
// O+=P*V both via bf16 WMMA; P goes through a per-wave LDS buffer to convert
// C-layout -> A-layout.
// ---------------------------------------------------------------------------
__global__ __launch_bounds__(128) void flash_attn_kernel(
    const bf16_t* __restrict__ qb, const bf16_t* __restrict__ kb,
    const bf16_t* __restrict__ vT, bf16_t* __restrict__ attn_out) {
  __shared__ __align__(16) bf16_t P[4][16 * 32];

  const int lane = threadIdx.x & 31, wave = threadIdx.x >> 5;
  const int lm = lane & 15, half = lane >> 4, h8 = half * 8;
  const int bh = blockIdx.y;
  const int q0 = blockIdx.x * 64 + wave * 16;

  const bf16_t* qB = qb + (size_t)bh * N_ * DH_;
  const bf16_t* kB = kb + (size_t)bh * N_ * DH_;
  const bf16_t* vB = vT + (size_t)bh * DH_ * N_;
  bf16_t* Pw = P[wave];

  // Q fragments for this wave's 16 rows (kept in registers for whole loop).
  const bf16_t* qrow = qB + (size_t)(q0 + lm) * DH_;
  v16bf qa0 = make_frag(qrow + h8,      qrow + 16 + h8);
  v16bf qa1 = make_frag(qrow + 32 + h8, qrow + 48 + h8);

  v8f o[4];
#pragma unroll
  for (int ot = 0; ot < 4; ++ot) o[ot] = zero8();
  float mrow[8], lrow[8];
#pragma unroll
  for (int r = 0; r < 8; ++r) { mrow[r] = -3.0e38f; lrow[r] = 0.f; }

  const float scale = 0.125f;  // 1/sqrt(64)
  const int kend = q0 + 16;    // causal: keys < kend

  for (int kk = 0; kk < kend; kk += 32) {
    // --- S = Q * K^T for keys [kk, kk+32) -> two 16x16 score tiles ---
    v8f s0t, s1t;
    {
      const bf16_t* krow = kB + (size_t)(kk + lm) * DH_;
      v8f z = zero8();
      z   = wmma_bf16(qa0, make_frag(krow + h8,      krow + 16 + h8), z);
      s0t = wmma_bf16(qa1, make_frag(krow + 32 + h8, krow + 48 + h8), z);
    }
    {
      const bf16_t* krow = kB + (size_t)(kk + 16 + lm) * DH_;
      v8f z = zero8();
      z   = wmma_bf16(qa0, make_frag(krow + h8,      krow + 16 + h8), z);
      s1t = wmma_bf16(qa1, make_frag(krow + 32 + h8, krow + 48 + h8), z);
    }

    // --- scale, causal mask, online softmax update ---
#pragma unroll
    for (int r = 0; r < 8; ++r) {
      int qidx = q0 + r + 8 * half;
      float a = s0t[r] * scale; if (kk + lm      > qidx) a = -3.0e38f;
      float b = s1t[r] * scale; if (kk + 16 + lm > qidx) b = -3.0e38f;
      // row max across the 16 lanes of this half-group
      float v = fmaxf(a, b);
      v = fmaxf(v, __shfl_xor(v, 1));
      v = fmaxf(v, __shfl_xor(v, 2));
      v = fmaxf(v, __shfl_xor(v, 4));
      v = fmaxf(v, __shfl_xor(v, 8));
      float mnew = fmaxf(mrow[r], v);
      float p0 = __expf(a - mnew);
      float p1 = __expf(b - mnew);
      Pw[(r + 8 * half) * 32 + lm]      = (bf16_t)p0;
      Pw[(r + 8 * half) * 32 + 16 + lm] = (bf16_t)p1;
      float ps = p0 + p1;
      ps += __shfl_xor(ps, 1);
      ps += __shfl_xor(ps, 2);
      ps += __shfl_xor(ps, 4);
      ps += __shfl_xor(ps, 8);
      float alpha = __expf(mrow[r] - mnew);
      lrow[r] = lrow[r] * alpha + ps;
      mrow[r] = mnew;
#pragma unroll
      for (int ot = 0; ot < 4; ++ot) o[ot][r] *= alpha;
    }

    // LDS round-trip: ensure probability stores landed before re-reading in
    // A-fragment layout (same-wave DS ops are in-order; barrier keeps the
    // compiler honest).
    asm volatile("s_wait_dscnt 0" ::: "memory");

    // --- O += P * V ---
    v16bf pf = make_frag(&Pw[lm * 32 + h8], &Pw[lm * 32 + 16 + h8]);
#pragma unroll
    for (int ot = 0; ot < 4; ++ot) {
      const bf16_t* vrow = vB + (size_t)(ot * 16 + lm) * N_ + kk;
      o[ot] = wmma_bf16(pf, make_frag(vrow + h8, vrow + 16 + h8), o[ot]);
    }
  }

  // --- normalize and store to [b, n, d] bf16 for the output projection ---
  int b = bh >> 4, hh = bh & 15;
#pragma unroll
  for (int ot = 0; ot < 4; ++ot) {
#pragma unroll
    for (int r = 0; r < 8; ++r) {
      int row = q0 + r + 8 * half;
      float val = o[ot][r] / lrow[r];
      attn_out[((size_t)(b * N_ + row)) * D_ + hh * DH_ + ot * 16 + lm] =
          (bf16_t)val;
    }
  }
}

// ---------------------------------------------------------------------------
// Launch
// ---------------------------------------------------------------------------
extern "C" void kernel_launch(void* const* d_in, const int* in_sizes, int n_in,
                              void* d_out, int out_size, void* d_ws,
                              size_t ws_size, hipStream_t stream) {
  (void)in_sizes; (void)n_in; (void)out_size; (void)ws_size;
  const float* x    = (const float*)d_in[0];  // (2,2048,1024)
  const float* wqkv = (const float*)d_in[1];  // (1024,3072)
  const float* wo   = (const float*)d_in[2];  // (1024,1024)
  const float* bl   = (const float*)d_in[3];  // (16,5)
  float* out = (float*)d_out;                 // (2,2048,1024) fp32

  char* ws = (char*)d_ws;
  size_t off = 0;
  auto wsalloc = [&](size_t bytes) -> void* {
    void* p = ws + off;
    off += (bytes + 255) & ~(size_t)255;
    return p;
  };
  bf16_t* xb    = (bf16_t*)wsalloc((size_t)M_TOT * D_ * 2);       // 8 MB
  bf16_t* wqkvT = (bf16_t*)wsalloc((size_t)N3_ * D_ * 2);         // 6 MB
  bf16_t* woT   = (bf16_t*)wsalloc((size_t)D_ * D_ * 2);          // 2 MB
  float*  cosT  = (float*)wsalloc((size_t)H_ * N_ * 32 * 4);      // 4 MB
  float*  sinT  = (float*)wsalloc((size_t)H_ * N_ * 32 * 4);      // 4 MB
  bf16_t* qbuf  = (bf16_t*)wsalloc((size_t)B_ * H_ * N_ * DH_ * 2); // 8 MB
  bf16_t* kbuf  = (bf16_t*)wsalloc((size_t)B_ * H_ * N_ * DH_ * 2); // 8 MB
  bf16_t* vTb   = (bf16_t*)wsalloc((size_t)B_ * H_ * DH_ * N_ * 2); // 8 MB
  bf16_t* ao    = (bf16_t*)wsalloc((size_t)M_TOT * D_ * 2);       // 8 MB

  {
    int n = M_TOT * D_;
    f32_to_bf16_kernel<<<(n + 255) / 256, 256, 0, stream>>>(x, xb, n);
  }
  transpose_to_bf16_kernel<<<(D_ * N3_ + 255) / 256, 256, 0, stream>>>(
      wqkv, wqkvT, D_, N3_);
  transpose_to_bf16_kernel<<<(D_ * D_ + 255) / 256, 256, 0, stream>>>(
      wo, woT, D_, D_);
  {
    int n = H_ * N_ * 32;
    euler_cossin_kernel<<<(n + 255) / 256, 256, 0, stream>>>(bl, cosT, sinT);
  }

  // QKV projection + fused RoPE scatter
  gemm_wmma_kernel<0><<<dim3(N3_ / 128, M_TOT / 128), 256, 0, stream>>>(
      xb, wqkvT, M_TOT, N3_, D_, nullptr, qbuf, kbuf, vTb, cosT, sinT);

  // Causal flash attention
  flash_attn_kernel<<<dim3(N_ / 64, B_ * H_), 128, 0, stream>>>(
      qbuf, kbuf, vTb, ao);

  // Output projection -> fp32 d_out
  gemm_wmma_kernel<1><<<dim3(D_ / 128, M_TOT / 128), 256, 0, stream>>>(
      ao, woT, M_TOT, D_, D_, out, nullptr, nullptr, nullptr, nullptr,
      nullptr);
}
